// ScaledEuclidDistanceAttention_32615981646584
// MI455X (gfx1250) — matrix operand order repaired and checked by reading, not compiled, not run on hardware
//
#include <hip/hip_runtime.h>

typedef __bf16 bf16_t;
typedef bf16_t v16bf __attribute__((ext_vector_type(16)));
typedef bf16_t v8bf  __attribute__((ext_vector_type(8)));
typedef float  v8f   __attribute__((ext_vector_type(8)));

#define S_LEN   2048
#define D_MODEL 1024
#define DH      64
#define WAVES   8
#define BLK_Q   128   // 8 waves * 16 q-rows
#define KT      64    // k columns per iteration

__device__ __forceinline__ float fast_rcp(float x)  { return __builtin_amdgcn_rcpf(x); }
__device__ __forceinline__ float fast_sqrt(float x) { return __builtin_amdgcn_sqrtf(x); }

__global__ __launch_bounds__(256)
void euclid_attn_kernel(const float* __restrict__ qg,
                        const float* __restrict__ kg,
                        const float* __restrict__ vg,
                        const int*   __restrict__ maskg,
                        float* __restrict__ outg)
{
    __shared__ __align__(32) bf16_t lds_q[WAVES][16][DH];   // 16 KB
    __shared__ __align__(32) bf16_t lds_k[KT][DH];          //  8 KB
    __shared__ __align__(32) bf16_t lds_vt[DH][KT];         //  8 KB (V transposed)
    __shared__ __align__(32) bf16_t lds_p[WAVES][16][KT];   // 16 KB
    __shared__ float lds_k2[KT];

    const int tid  = threadIdx.x;
    const int lane = tid & 31;
    const int wave = tid >> 5;
    const int half = lane >> 4;     // lane half: selects K-chunk / M-group
    const int n15  = lane & 15;

    const int bq    = blockIdx.x;
    const int qtile = bq & 15;          // S/128 = 16 q tiles
    const int h     = (bq >> 4) & 15;   // 16 heads
    const int b     = bq >> 8;          // 2 batches
    const int qbase = qtile * BLK_Q + wave * 16;
    const size_t head_off = (size_t)h * DH;

    // ---- per-wave: load Q tile (fp32 -> bf16 into LDS), fp32 row norms ----
    float rowsum;
    {
        const int row = lane >> 1;
        const int c0  = (lane & 1) * 32;
        const float* qp = qg + ((size_t)(b * S_LEN + qbase + row)) * D_MODEL + head_off + c0;
        float s2 = 0.f;
#pragma unroll
        for (int j = 0; j < 4; ++j) {           // 4 chunks of 8 floats
            float4 f0 = ((const float4*)qp)[2*j];
            float4 f1 = ((const float4*)qp)[2*j + 1];
            s2 += f0.x*f0.x + f0.y*f0.y + f0.z*f0.z + f0.w*f0.w;
            s2 += f1.x*f1.x + f1.y*f1.y + f1.z*f1.z + f1.w*f1.w;
            v8bf pk;
            pk[0]=(bf16_t)f0.x; pk[1]=(bf16_t)f0.y; pk[2]=(bf16_t)f0.z; pk[3]=(bf16_t)f0.w;
            pk[4]=(bf16_t)f1.x; pk[5]=(bf16_t)f1.y; pk[6]=(bf16_t)f1.z; pk[7]=(bf16_t)f1.w;
            *(v8bf*)&lds_q[wave][row][c0 + 8*j] = pk;
        }
        s2 += __shfl_xor(s2, 1, 32);   // combine the two half-rows
        rowsum = s2;                    // lanes 2r,2r+1 hold q2 of row r (r=lane>>1)
    }
    // q2 for the 8 rows this lane's half covers in C/D layout (M = r + 8*half)
    float q2r[8];
#pragma unroll
    for (int r = 0; r < 8; ++r)
        q2r[r] = __shfl(rowsum, (half << 4) + 2 * r, 32);

    __builtin_amdgcn_wave_barrier();

    // ---- Q A-fragments: A[M][K] with M=n15; chunks at K=8*half and 16+8*half ----
    v16bf qf[2];
#pragma unroll
    for (int ks = 0; ks < 2; ++ks) {
        const bf16_t* base = &lds_q[wave][n15][ks * 32 + 8 * half];
        v8bf lo = *(const v8bf*)base;
        v8bf hi = *(const v8bf*)(base + 16);
#pragma unroll
        for (int i = 0; i < 8; ++i) { qf[ks][i] = lo[i]; qf[ks][i + 8] = hi[i]; }
    }

    v8f O[4];
#pragma unroll
    for (int dt = 0; dt < 4; ++dt) O[dt] = v8f{0.f,0.f,0.f,0.f,0.f,0.f,0.f,0.f};
    float mrun[8], lrun[8];
#pragma unroll
    for (int r = 0; r < 8; ++r) { mrun[r] = -3.0e38f; lrun[r] = 0.f; }

    // =============== online-softmax loop over K tiles ===============
#pragma unroll 1
    for (int kb = 0; kb < S_LEN; kb += KT) {
        __syncthreads();   // previous tile fully consumed
        if (tid < 128) {   // waves 0-3: K tile + fp32 k2
            const int row = tid >> 1, c0 = (tid & 1) * 32;
            const float* kp = kg + ((size_t)(b * S_LEN + kb + row)) * D_MODEL + head_off + c0;
            float s2 = 0.f;
#pragma unroll
            for (int j = 0; j < 4; ++j) {
                float4 f0 = ((const float4*)kp)[2*j];
                float4 f1 = ((const float4*)kp)[2*j + 1];
                s2 += f0.x*f0.x + f0.y*f0.y + f0.z*f0.z + f0.w*f0.w;
                s2 += f1.x*f1.x + f1.y*f1.y + f1.z*f1.z + f1.w*f1.w;
                v8bf pk;
                pk[0]=(bf16_t)f0.x; pk[1]=(bf16_t)f0.y; pk[2]=(bf16_t)f0.z; pk[3]=(bf16_t)f0.w;
                pk[4]=(bf16_t)f1.x; pk[5]=(bf16_t)f1.y; pk[6]=(bf16_t)f1.z; pk[7]=(bf16_t)f1.w;
                *(v8bf*)&lds_k[row][c0 + 8*j] = pk;
            }
            s2 += __shfl_xor(s2, 1, 32);
            if ((tid & 1) == 0) lds_k2[row] = s2;
            if (kb + KT < S_LEN) __builtin_prefetch(kp + (size_t)KT * D_MODEL, 0, 1);
        } else {           // waves 4-7: V tile, stored transposed VT[d][kcol]
            const int row = (tid - 128) >> 1, c0 = (tid & 1) * 32;
            const float* vp = vg + ((size_t)(b * S_LEN + kb + row)) * D_MODEL + head_off + c0;
#pragma unroll
            for (int j = 0; j < 8; ++j) {
                float4 f = ((const float4*)vp)[j];
                lds_vt[c0 + 4*j + 0][row] = (bf16_t)f.x;
                lds_vt[c0 + 4*j + 1][row] = (bf16_t)f.y;
                lds_vt[c0 + 4*j + 2][row] = (bf16_t)f.z;
                lds_vt[c0 + 4*j + 3][row] = (bf16_t)f.w;
            }
            if (kb + KT < S_LEN) __builtin_prefetch(vp + (size_t)KT * D_MODEL, 0, 1);
        }
        __syncthreads();

        // ---- GEMM1: scores = Q . K^T for four 16x16 tiles ----
        float sc[4][8];
#pragma unroll
        for (int kt = 0; kt < 4; ++kt) {
            v8f c = v8f{0.f,0.f,0.f,0.f,0.f,0.f,0.f,0.f};
#pragma unroll
            for (int ks = 0; ks < 2; ++ks) {
                // B[K][N]: N = n15 (key column), chunks at K = 8*half and 16+8*half
                const bf16_t* bp = &lds_k[kt * 16 + n15][ks * 32 + 8 * half];
                v8bf blo = *(const v8bf*)bp;
                v8bf bhi = *(const v8bf*)(bp + 16);
                v16bf bf;
#pragma unroll
                for (int i = 0; i < 8; ++i) { bf[i] = blo[i]; bf[i + 8] = bhi[i]; }
                c = __builtin_amdgcn_wmma_f32_16x16x32_bf16(
                        false, qf[ks], false, bf, (short)0, c, false, false);
            }
            const float k2v = lds_k2[kt * 16 + n15];
            const int kcol  = kb + kt * 16 + n15;
#pragma unroll
            for (int r = 0; r < 8; ++r) {
                const int qrow = qbase + r + 8 * half;
                float d2 = q2r[r] + k2v - 2.0f * c[r];
                d2 = fmaxf(d2, 0.0f);
                float dist = fast_sqrt(d2) * 0.125f;          // dh^-0.5
                float lg = fast_rcp(dist + 1e-9f);
                lg = fmaf((float)maskg[(size_t)qrow * S_LEN + kcol], -1.0e9f, lg);
                sc[kt][r] = lg;
            }
        }

        // ---- online softmax update (row stats across 16 lanes of the half) ----
        float cf[8];
#pragma unroll
        for (int r = 0; r < 8; ++r) {
            float tm = fmaxf(fmaxf(sc[0][r], sc[1][r]), fmaxf(sc[2][r], sc[3][r]));
            tm = fmaxf(tm, __shfl_xor(tm, 1, 32));
            tm = fmaxf(tm, __shfl_xor(tm, 2, 32));
            tm = fmaxf(tm, __shfl_xor(tm, 4, 32));
            tm = fmaxf(tm, __shfl_xor(tm, 8, 32));
            const float mnew = fmaxf(mrun[r], tm);
            const float corr = __expf(mrun[r] - mnew);
            mrun[r] = mnew;
            float ps = 0.f;
#pragma unroll
            for (int kt = 0; kt < 4; ++kt) {
                float p = __expf(sc[kt][r] - mnew);
                sc[kt][r] = p;
                ps += p;
            }
            ps += __shfl_xor(ps, 1, 32);
            ps += __shfl_xor(ps, 2, 32);
            ps += __shfl_xor(ps, 4, 32);
            ps += __shfl_xor(ps, 8, 32);
            lrun[r] = lrun[r] * corr + ps;
            cf[r] = corr;
        }
#pragma unroll
        for (int dt = 0; dt < 4; ++dt)
#pragma unroll
            for (int r = 0; r < 8; ++r) O[dt][r] *= cf[r];

        // ---- P (bf16) -> wave-private LDS, reload as A-fragments ----
#pragma unroll
        for (int kt = 0; kt < 4; ++kt)
#pragma unroll
            for (int r = 0; r < 8; ++r)
                lds_p[wave][r + 8 * half][kt * 16 + n15] = (bf16_t)sc[kt][r];
        __builtin_amdgcn_wave_barrier();

        v16bf pf[2];
#pragma unroll
        for (int ks = 0; ks < 2; ++ks) {
            const bf16_t* base = &lds_p[wave][n15][ks * 32 + 8 * half];
            v8bf lo = *(const v8bf*)base;
            v8bf hi = *(const v8bf*)(base + 16);
#pragma unroll
            for (int i = 0; i < 8; ++i) { pf[ks][i] = lo[i]; pf[ks][i + 8] = hi[i]; }
        }

        // ---- GEMM2: O += P . V  (B from transposed V tile, contiguous) ----
#pragma unroll
        for (int dt = 0; dt < 4; ++dt) {
#pragma unroll
            for (int ks = 0; ks < 2; ++ks) {
                const bf16_t* vb = &lds_vt[dt * 16 + n15][ks * 32 + 8 * half];
                v8bf vlo = *(const v8bf*)vb;
                v8bf vhi = *(const v8bf*)(vb + 16);
                v16bf vf;
#pragma unroll
                for (int i = 0; i < 8; ++i) { vf[i] = vlo[i]; vf[i + 8] = vhi[i]; }
                O[dt] = __builtin_amdgcn_wmma_f32_16x16x32_bf16(
                            false, pf[ks], false, vf, (short)0, O[dt], false, false);
            }
        }
    }

    // ---- epilogue: normalize and store fp32 ----
#pragma unroll
    for (int r = 0; r < 8; ++r) lrun[r] = fast_rcp(lrun[r]);
#pragma unroll
    for (int dt = 0; dt < 4; ++dt) {
#pragma unroll
        for (int r = 0; r < 8; ++r) {
            const int qrow = qbase + r + 8 * half;
            const size_t o = ((size_t)(b * S_LEN + qrow)) * D_MODEL + head_off + dt * 16 + n15;
            outg[o] = O[dt][r] * lrun[r];
        }
    }
}

extern "C" void kernel_launch(void* const* d_in, const int* in_sizes, int n_in,
                              void* d_out, int out_size, void* d_ws, size_t ws_size,
                              hipStream_t stream) {
    (void)in_sizes; (void)n_in; (void)d_ws; (void)ws_size; (void)out_size;
    const float* q    = (const float*)d_in[0];
    const float* k    = (const float*)d_in[1];
    const float* v    = (const float*)d_in[2];
    const int*   mask = (const int*)d_in[3];
    // d_in[4] = num_heads (16), hardcoded in the kernel tiling
    float* out = (float*)d_out;

    const int blocks = 2 /*B*/ * 16 /*H*/ * (S_LEN / BLK_Q); // 512
    euclid_attn_kernel<<<blocks, 256, 0, stream>>>(q, k, v, mask, out);
}
